// LearnedEdge_9955734192569
// MI455X (gfx1250) — matrix operand order repaired and checked by reading, not compiled, not run on hardware
//
#include <hip/hip_runtime.h>

// ---- trivial vector types (union-safe, ext_vector) ----
typedef __attribute__((ext_vector_type(16))) __bf16        v16bf;
typedef __attribute__((ext_vector_type(8)))  float         v8f;
typedef __attribute__((ext_vector_type(4)))  float         f32x4;
typedef __attribute__((ext_vector_type(4)))  unsigned int  u32x4;

#define NN 512      // T + taus
#define TT 256      // T
#define DD 64       // feature dim
#define EB 98176    // edges per batch

__device__ __forceinline__ unsigned short f2bf(float f) {
    unsigned u = __float_as_uint(f);
    u += 0x7FFFu + ((u >> 16) & 1u);        // round-to-nearest-even
    return (unsigned short)(u >> 16);
}

// pack two f32 -> bf16x2 (integer-only, avoids b16 subreg shuffling)
__device__ __forceinline__ unsigned pack_bf16(float a, float b) {
    unsigned ua = __float_as_uint(a), ub = __float_as_uint(b);
    ua += 0x7FFFu + ((ua >> 16) & 1u);
    ub += 0x7FFFu + ((ub >> 16) & 1u);
    return (ua >> 16) | (ub & 0xFFFF0000u);
}

// xor-butterfly lane exchange: DPP row_xmask for masks <16, v_permlanex16 for 16.
// Pure VALU (no ds_bpermute / s_wait_dscnt), leaving the LDS port to WMMA staging.
template <int M>
__device__ __forceinline__ float lane_xor(float v) {
#if __has_builtin(__builtin_amdgcn_update_dpp) && __has_builtin(__builtin_amdgcn_permlanex16)
    if constexpr (M == 16) {
        return __int_as_float(__builtin_amdgcn_permlanex16(
            __float_as_int(v), __float_as_int(v), 0x76543210u, 0xfedcba98u,
            true, false));
    } else {
        return __int_as_float(__builtin_amdgcn_update_dpp(
            0, __float_as_int(v), 0x160 | M, 0xf, 0xf, true));
    }
#else
    return __shfl_xor(v, M);
#endif
}

__device__ __forceinline__ float red16_add(float v) {   // sum over 16-lane half
    v += lane_xor<1>(v); v += lane_xor<2>(v);
    v += lane_xor<4>(v); v += lane_xor<8>(v);
    return v;
}

// Kernel 1: A[b,i] = nodes[b,i] @ W1[:64] + b1 ; C[b,i] = nodes[b,i] @ W1[64:]
__global__ void precompute_ac(const float* __restrict__ nodes,
                              const float* __restrict__ W1,
                              const float* __restrict__ b1,
                              float* __restrict__ A, float* __restrict__ C) {
    __shared__ float nr[DD];
    const int row = blockIdx.x;             // b*N + i
    const int c   = threadIdx.x;            // 0..63
    nr[c] = nodes[(size_t)row * DD + c];
    __syncthreads();
    float a = b1[c], cc = 0.f;
    #pragma unroll 8
    for (int k = 0; k < DD; ++k) {
        const float nv = nr[k];
        a  += nv * W1[k * DD + c];
        cc += nv * W1[(DD + k) * DD + c];
    }
    A[(size_t)row * DD + c] = a;
    C[(size_t)row * DD + c] = cc;
}

// Kernel 2: fused edge-MLP (bf16 WMMA) + segment gumbel-softmax + dense row write.
// One block per (b, sink); 256 threads = 8 waves; each wave owns 16-src tiles.
__global__ void __launch_bounds__(256)
edge_adj_kernel(const float* __restrict__ A,  const float* __restrict__ C,
                const float* __restrict__ gumbel,
                const float* __restrict__ g1, const float* __restrict__ be1,
                const float* __restrict__ W2, const float* __restrict__ b2,
                const float* __restrict__ g2, const float* __restrict__ be2,
                const float* __restrict__ W3, const float* __restrict__ b3,
                const float* __restrict__ tau, float* __restrict__ out) {
    const int bs   = blockIdx.x;            // b*N + sink
    const int sink = bs & (NN - 1);
    const int b    = bs >> 9;
    const int tid  = threadIdx.x;
    float* orow = out + (size_t)bs * NN;

    if (sink < TT) {                        // no incoming edges: zero row
        orow[tid]       = 0.f;
        orow[tid + 256] = 0.f;
        return;
    }

    const int lane = tid & 31;
    const int wv   = tid >> 5;
    const int half = (lane >= 16) ? 1 : 0;

    __shared__ __align__(16) unsigned short h1s[8][16][DD];  // per-wave bf16 h1 tile
    __shared__ float asld[DD], sg1[DD], sbe1[DD];
    __shared__ float ys[NN];
    __shared__ float redA[8], redB[8];

    if (tid < DD) {
        asld[tid] = A[(size_t)bs * DD + tid];
        sg1[tid]  = g1[tid];
        sbe1[tid] = be1[tid];
    }
    __syncthreads();

    // loop-invariant per-lane column params (n = nb*16 + lane%16)
    float b2r[4], gw[4];
    float gpart = 0.f, bpart = 0.f;
    #pragma unroll
    for (int nb = 0; nb < 4; ++nb) {
        const int n = nb * 16 + (lane & 15);
        const float w3n = W3[n];
        b2r[nb] = b2[n];
        gw[nb]  = g2[n] * w3n;              // g2*w3 fused weight
        gpart  += gw[nb];
        bpart  += be2[n] * w3n;
    }
    // block-uniform LN2 constants: G = sum(g2*w3), Kb = sum(be2*w3)
    const float Gs = red16_add(gpart);
    const float Kb = red16_add(bpart) + b3[0];

    // W2 as 8 bf16 WMMA B-tiles (B layout: lane holds col n=lane%16, K = kb*32 + half*16 + e)
    v16bf Bt[2][4];
    #pragma unroll
    for (int kb = 0; kb < 2; ++kb)
        #pragma unroll
        for (int nb = 0; nb < 4; ++nb) {
            union { unsigned short us[16]; v16bf v; } u;
            const int K0 = kb * 32 + half * 16;
            const int n  = nb * 16 + (lane & 15);
            #pragma unroll
            for (int e = 0; e < 16; ++e)
                u.us[e] = f2bf(W2[(K0 + e) * DD + n]);
            Bt[kb][nb] = u.v;
        }

    const int ntiles  = (sink + 15) >> 4;
    const int r       = lane & 15;          // tile row this lane computes h1 for
    const int colbase = half * 32;          // which half of the 64 features

    for (int t = wv; t < ntiles; t += 8) {
        const int tbase = t << 4;
        const int src   = tbase + r;        // always < 512 (valid memory)
        const float* crow = C + ((size_t)(b * NN + src)) * DD + colbase;
        if (t + 8 < ntiles)                  // prefetch next tile's C rows for this wave
            __builtin_prefetch(crow + 128 * DD, 0, 3);   // -> global_prefetch_b8

        // ---- h1 = LayerNorm(relu(A_sink + C_src)), 32 features per lane ----
        float tv[32]; float s = 0.f, ss = 0.f;
        #pragma unroll
        for (int j = 0; j < 32; j += 4) {
            const f32x4 cv = *(const f32x4*)(crow + j);
            const f32x4 av = *(const f32x4*)(&asld[colbase + j]);
            const float x0 = fmaxf(av.x + cv.x, 0.f);
            const float x1 = fmaxf(av.y + cv.y, 0.f);
            const float x2 = fmaxf(av.z + cv.z, 0.f);
            const float x3 = fmaxf(av.w + cv.w, 0.f);
            tv[j] = x0; tv[j+1] = x1; tv[j+2] = x2; tv[j+3] = x3;
            s  += x0 + x1 + x2 + x3;
            ss += x0*x0 + x1*x1 + x2*x2 + x3*x3;
        }
        s  += lane_xor<16>(s);              // combine the two half-rows (permlanex16)
        ss += lane_xor<16>(ss);
        const float mean = s * (1.f / 64.f);
        const float inv  = rsqrtf(ss * (1.f / 64.f) - mean * mean + 1e-5f);

        unsigned pk[16];
        #pragma unroll
        for (int j = 0; j < 32; j += 4) {
            const f32x4 gv = *(const f32x4*)(&sg1[colbase + j]);
            const f32x4 bv = *(const f32x4*)(&sbe1[colbase + j]);
            const float h0 = (tv[j]   - mean) * inv * gv.x + bv.x;
            const float h1 = (tv[j+1] - mean) * inv * gv.y + bv.y;
            const float h2 = (tv[j+2] - mean) * inv * gv.z + bv.z;
            const float h3 = (tv[j+3] - mean) * inv * gv.w + bv.w;
            pk[(j >> 1)]     = pack_bf16(h0, h1);
            pk[(j >> 1) + 1] = pack_bf16(h2, h3);
        }
        u32x4* dst = (u32x4*)&h1s[wv][r][colbase];
        #pragma unroll
        for (int q = 0; q < 4; ++q) dst[q] = ((const u32x4*)pk)[q];
        // same-wave LDS RAW: DS ops execute in order within a wave

        // ---- H2pre(16x64) = H1(16x64) @ W2(64x64): 8x v_wmma_f32_16x16x32_bf16 ----
        v8f acc[4] = { {0,0,0,0,0,0,0,0}, {0,0,0,0,0,0,0,0},
                       {0,0,0,0,0,0,0,0}, {0,0,0,0,0,0,0,0} };
        #pragma unroll
        for (int kb = 0; kb < 2; ++kb) {
            // A layout: lanes 0-15 hold K {base..base+15}, lanes 16-31 shifted +16
            const int kbase = kb * 32 + half * 16;
            union { u32x4 q[2]; v16bf v; } au;
            au.q[0] = *(const u32x4*)&h1s[wv][r][kbase];
            au.q[1] = *(const u32x4*)&h1s[wv][r][kbase + 8];
            #pragma unroll
            for (int nb = 0; nb < 4; ++nb)
                acc[nb] = __builtin_amdgcn_wmma_f32_16x16x32_bf16(
                    false, au.v, false, Bt[kb][nb], (short)0, acc[nb], false, false);
        }

        // ---- relu + LayerNorm + dot(W3), algebraically fused:
        //      logit = inv*(Sg - mean*G) + Kb, needing only {Sx, Sxx, Sg} per row ----
        #pragma unroll
        for (int e = 0; e < 8; ++e) {
            float p1 = 0.f, p2 = 0.f, pg = 0.f;
            #pragma unroll
            for (int nb = 0; nb < 4; ++nb) {
                const float x = fmaxf(acc[nb][e] + b2r[nb], 0.f);
                p1 += x; p2 += x * x; pg += x * gw[nb];
            }
            p1 = red16_add(p1);             // DPP row_xmask butterflies
            p2 = red16_add(p2);
            pg = red16_add(pg);
            const float mean2 = p1 * (1.f / 64.f);
            const float inv2  = rsqrtf(p2 * (1.f / 64.f) - mean2 * mean2 + 1e-5f);
            if ((lane & 15) == 0)           // D layout: this row m = half*8 + e
                ys[tbase + half * 8 + e] = inv2 * (pg - mean2 * Gs) + Kb;
        }
    }
    __syncthreads();

    // ---- segment gumbel-softmax over src in [0, sink), threshold at 1/6 ----
    const float tc   = fminf(fmaxf(tau[0], 0.001f), 5.0f);
    const float itau = 1.f / tc;
    const int   gb   = b * EB + ((sink * (sink - 1)) >> 1) - 32640;  // analytic edge base

    float yv0 = -3.0e38f, yv1 = -3.0e38f;
    if (tid < sink)       yv0 = (ys[tid]       + gumbel[gb + tid])       * itau;
    if (tid + 256 < sink) yv1 = (ys[tid + 256] + gumbel[gb + tid + 256]) * itau;

    float lmax = fmaxf(yv0, yv1);
    lmax = fmaxf(lmax, lane_xor<1>(lmax));
    lmax = fmaxf(lmax, lane_xor<2>(lmax));
    lmax = fmaxf(lmax, lane_xor<4>(lmax));
    lmax = fmaxf(lmax, lane_xor<8>(lmax));
    lmax = fmaxf(lmax, lane_xor<16>(lmax));
    if (lane == 0) redA[wv] = lmax;
    __syncthreads();
    float bmax = redA[0];
    #pragma unroll
    for (int k = 1; k < 8; ++k) bmax = fmaxf(bmax, redA[k]);

    float lsum = 0.f;
    if (tid < sink)       lsum += expf(yv0 - bmax);
    if (tid + 256 < sink) lsum += expf(yv1 - bmax);
    lsum = red16_add(lsum);
    lsum += lane_xor<16>(lsum);
    if (lane == 0) redB[wv] = lsum;
    __syncthreads();
    float bsum = 0.f;
    #pragma unroll
    for (int k = 0; k < 8; ++k) bsum += redB[k];

    // v > 1/6  <=>  y > max + ln(sum) - ln 6 ; kept edges write exactly 1.0
    const float thr = bmax + logf(bsum) - 1.791759469228055f;
    orow[tid]       = (tid < sink       && yv0 > thr) ? 1.f : 0.f;
    orow[tid + 256] = (tid + 256 < sink && yv1 > thr) ? 1.f : 0.f;
}

extern "C" void kernel_launch(void* const* d_in, const int* in_sizes, int n_in,
                              void* d_out, int out_size, void* d_ws, size_t ws_size,
                              hipStream_t stream) {
    (void)in_sizes; (void)n_in; (void)out_size; (void)ws_size;
    const float* nodes  = (const float*)d_in[0];
    /* d_in[1] = edge_idx: unused, indices are analytic */
    const float* gumbel = (const float*)d_in[2];
    const float* W1     = (const float*)d_in[3];
    const float* b1     = (const float*)d_in[4];
    const float* g1     = (const float*)d_in[5];
    const float* be1    = (const float*)d_in[6];
    const float* W2     = (const float*)d_in[7];
    const float* b2     = (const float*)d_in[8];
    const float* g2     = (const float*)d_in[9];
    const float* be2    = (const float*)d_in[10];
    const float* W3     = (const float*)d_in[11];
    const float* b3     = (const float*)d_in[12];
    const float* tau    = (const float*)d_in[13];
    float* out = (float*)d_out;

    float* Aws = (float*)d_ws;              // [B*N, 64] f32 (1 MB)
    float* Cws = Aws + 8 * NN * DD;         // [B*N, 64] f32 (1 MB)

    precompute_ac<<<8 * NN, DD, 0, stream>>>(nodes, W1, b1, Aws, Cws);
    edge_adj_kernel<<<8 * NN, 256, 0, stream>>>(Aws, Cws, gumbel, g1, be1,
                                                W2, b2, g2, be2, W3, b3, tau, out);
}
